// Attention_48095043781121
// MI455X (gfx1250) — matrix operand order repaired and checked by reading, not compiled, not run on hardware
//
#include <hip/hip_runtime.h>
#include <hip/hip_bf16.h>
#include <stdint.h>

typedef __attribute__((ext_vector_type(16))) __bf16 v16bf;
typedef __attribute__((ext_vector_type(8)))  __bf16 v8bf;
typedef __attribute__((ext_vector_type(8)))  float  v8f;

// Problem constants
static constexpr int kDIM = 512;
static constexpr int kH   = 8;
static constexpr int kDH  = 64;
static constexpr int kB   = 4;
static constexpr int kS   = 512;
static constexpr int kTOK = kB * kS;   // 2048

// ---------------------------------------------------------------------------
// WMMA helpers (CDNA5 bf16 16x16x32, fp32 accumulate)
// ---------------------------------------------------------------------------
__device__ __forceinline__ v8f wmma_bf16(v16bf a, v16bf b, v8f c) {
  return __builtin_amdgcn_wmma_f32_16x16x32_bf16(
      /*neg_a=*/false, a, /*neg_b=*/false, b,
      /*c_mod=*/(short)0, c, /*reuse_a=*/false, /*reuse_b=*/false);
}

// A-matrix fragment (16x32 bf16). Lane L holds row M = L&15.
// ISA layout: VGPR0-3 hold K = 8*(L>>4) + 0..7 ; VGPR4-7 hold K = 16 + 8*(L>>4) + 0..7.
__device__ __forceinline__ v16bf load_a_frag(const __bf16* row, int k0, int lane) {
  const int kb = k0 + ((lane >> 4) << 3);
  v8bf lo = *(const v8bf*)(row + kb);
  v8bf hi = *(const v8bf*)(row + kb + 16);
  v16bf a;
#pragma unroll
  for (int e = 0; e < 8; ++e) { a[e] = lo[e]; a[e + 8] = hi[e]; }
  return a;
}

// B-matrix fragment (32x16 bf16). Lane L holds column N = L&15;
// elements 0-7 carry K = k0 + 8*(L>>4) + 0..7, elements 8-15 carry K = k0 + 16 + 8*(L>>4) + 0..7.
// `col` points at the K-major data of this lane's column.
__device__ __forceinline__ v16bf load_b_frag(const __bf16* col, int k0, int lane) {
  const int kb = k0 + ((lane >> 4) << 3);
  v8bf lo = *(const v8bf*)(col + kb);
  v8bf hi = *(const v8bf*)(col + kb + 16);
  v16bf b;
#pragma unroll
  for (int e = 0; e < 8; ++e) { b[e] = lo[e]; b[e + 8] = hi[e]; }
  return b;
}

// ---------------------------------------------------------------------------
// Prep kernels
// ---------------------------------------------------------------------------
__global__ void k_convert_x(const float* __restrict__ src, __bf16* __restrict__ dst, int n) {
  int i = blockIdx.x * blockDim.x + threadIdx.x;
  if (i < n) dst[i] = (__bf16)src[i];
}

// Convert 512x512 fp32 row-major W[k][n] -> bf16 transposed WT[n][k]
__global__ void k_transpose_w(const float* __restrict__ s0, const float* __restrict__ s1,
                              const float* __restrict__ s2, const float* __restrict__ s3,
                              __bf16* __restrict__ d0, __bf16* __restrict__ d1,
                              __bf16* __restrict__ d2, __bf16* __restrict__ d3) {
  int i = blockIdx.x * blockDim.x + threadIdx.x;         // 0 .. 512*512-1
  const float* s = blockIdx.y == 0 ? s0 : blockIdx.y == 1 ? s1 : blockIdx.y == 2 ? s2 : s3;
  __bf16*      d = blockIdx.y == 0 ? d0 : blockIdx.y == 1 ? d1 : blockIdx.y == 2 ? d2 : d3;
  int k = i >> 9, n = i & 511;
  d[(size_t)n * kDIM + k] = (__bf16)s[i];
}

// Positional attention collapses: pos_attn[b,h,i,j] = softmax_j(-s[b,j,h]),
// s = (relu(pos@Wp1+bp1)@Wp2+bp2) @ Wh   (bh and the i-dependent term cancel
// in the softmax; shift invariance).
__global__ void k_pos(const float* __restrict__ pos, const float* __restrict__ Wp1,
                      const float* __restrict__ bp1, const float* __restrict__ Wp2,
                      const float* __restrict__ bp2, const float* __restrict__ Wh,
                      float* __restrict__ wpos) {
  __shared__ float buf[kS];
  const int bhid = blockIdx.x;            // 0..31  (b*8+h)
  const int b = bhid >> 3, h = bhid & 7;
  const int j = threadIdx.x;              // 0..511
  const float* pj = pos + ((size_t)b * kS + j) * 3;

  float hid[3];
#pragma unroll
  for (int i = 0; i < 3; ++i) {
    float t = bp1[i];
#pragma unroll
    for (int c = 0; c < 3; ++c) t += pj[c] * Wp1[c * 3 + i];
    hid[i] = fmaxf(t, 0.0f);
  }
  float s = 0.0f;
  for (int d = 0; d < 64; ++d) {
    float pd = bp2[d];
#pragma unroll
    for (int i = 0; i < 3; ++i) pd += hid[i] * Wp2[i * 64 + d];
    s += pd * Wh[d * kH + h];
  }
  float val = -s;
  buf[j] = val; __syncthreads();
  for (int st = 256; st > 0; st >>= 1) { if (j < st) buf[j] = fmaxf(buf[j], buf[j + st]); __syncthreads(); }
  float mx = buf[0]; __syncthreads();
  float e = __expf(val - mx);
  buf[j] = e; __syncthreads();
  for (int st = 256; st > 0; st >>= 1) { if (j < st) buf[j] += buf[j + st]; __syncthreads(); }
  wpos[(size_t)bhid * kS + j] = e / buf[0];
}

// ---------------------------------------------------------------------------
// QKV projection: [2048,512] @ [512,512] (x3), bf16 WMMA, fp32 accum.
// Writes q,k as bf16 [B,H,S,DH] and v transposed as bf16 [B,H,DH,S].
// grid = (8 n-strips, 16 m-groups, 3 matrices), block = 256 (8 waves).
// ---------------------------------------------------------------------------
__global__ void k_qkv_gemm(const __bf16* __restrict__ xb,
                           const __bf16* __restrict__ WqT, const __bf16* __restrict__ WkT,
                           const __bf16* __restrict__ WvT,
                           __bf16* __restrict__ qh, __bf16* __restrict__ kh,
                           __bf16* __restrict__ vT) {
  const int w = threadIdx.x >> 5, lane = threadIdx.x & 31;
  const int m0 = (blockIdx.y * 8 + w) * 16;
  const int n0 = blockIdx.x * 64;
  const int which = blockIdx.z;
  const __bf16* WT = which == 0 ? WqT : (which == 1 ? WkT : WvT);
  const __bf16* arow = xb + (size_t)(m0 + (lane & 15)) * kDIM;

  v8f acc[4] = {};
  for (int k0 = 0; k0 < kDIM; k0 += 32) {
    v16bf a = load_a_frag(arow, k0, lane);
#pragma unroll
    for (int t = 0; t < 4; ++t) {
      v16bf bfr = load_b_frag(WT + (size_t)(n0 + t * 16 + (lane & 15)) * kDIM, k0, lane);
      acc[t] = wmma_bf16(a, bfr, acc[t]);
    }
  }
  const int rofs = (lane >> 4) << 3;
#pragma unroll
  for (int t = 0; t < 4; ++t) {
#pragma unroll
    for (int v = 0; v < 8; ++v) {
      const int m = m0 + v + rofs;
      const int n = n0 + t * 16 + (lane & 15);
      const int b = m >> 9, s = m & 511, h = n >> 6, d = n & 63;
      const __bf16 val = (__bf16)acc[t][v];
      if (which == 0)      qh[((size_t)(b * kH + h) * kS + s) * kDH + d] = val;
      else if (which == 1) kh[((size_t)(b * kH + h) * kS + s) * kDH + d] = val;
      else                 vT[((size_t)(b * kH + h) * kDH + d) * kS + s] = val;
    }
  }
}

// ---------------------------------------------------------------------------
// Fused attention per (b,h, 16-query tile):
//   scores = q@k^T / 8  (WMMA) -> LDS -> softmax -> gate-combine with wpos
//   -> bf16 probs in LDS -> P@V via split-K WMMA -> aout bf16 [TOK, DIM]
// grid = (32 q-tiles, 32 bh), block = 256 (8 waves).
// ---------------------------------------------------------------------------
__global__ void k_attn(const __bf16* __restrict__ qh, const __bf16* __restrict__ kh,
                       const __bf16* __restrict__ vT, const float* __restrict__ wpos,
                       const float* __restrict__ gate, __bf16* __restrict__ aout) {
  __shared__ __align__(32) float  scores[16][kS];   // 32 KB
  __shared__ __align__(32) __bf16 pLDS[16][kS];     // 16 KB
  __shared__ float red[16][16];
  __shared__ float rowmax[16], rowsum[16];
  __shared__ float partial[4][16][16];              // split-K exchange

  const int tid = threadIdx.x;
  const int w = tid >> 5, lane = tid & 31;
  const int bhid = blockIdx.y;
  const int b = bhid >> 3, h = bhid & 7;
  const int q0 = blockIdx.x * 16;
  const int rofs = (lane >> 4) << 3;

  // ---- Phase 1: scores 16x512, wave w covers columns [w*64, w*64+64) ----
  {
    const __bf16* qrow = qh + ((size_t)bhid * kS + q0 + (lane & 15)) * kDH;
    const int jbase = w * 64;
    v8f acc[4] = {};
    for (int k0 = 0; k0 < kDH; k0 += 32) {
      v16bf a = load_a_frag(qrow, k0, lane);
#pragma unroll
      for (int t = 0; t < 4; ++t) {
        v16bf bfr = load_b_frag(kh + ((size_t)bhid * kS + jbase + t * 16 + (lane & 15)) * kDH,
                                k0, lane);
        acc[t] = wmma_bf16(a, bfr, acc[t]);
      }
    }
#pragma unroll
    for (int t = 0; t < 4; ++t)
#pragma unroll
      for (int v = 0; v < 8; ++v)
        scores[v + rofs][jbase + t * 16 + (lane & 15)] = acc[t][v] * 0.125f;  // 1/sqrt(64)
  }
  __syncthreads();

  // ---- Phase 2: row softmax + sigmoid-gate combine with positional weights ----
  {
    const float g = 1.0f / (1.0f + __expf(-gate[h]));
    const float* wp = wpos + (size_t)bhid * kS;
    const int r = tid >> 4, c = tid & 15;
    float mx = -3.0e38f;
    for (int i = 0; i < 32; ++i) mx = fmaxf(mx, scores[r][c + (i << 4)]);
    red[r][c] = mx; __syncthreads();
    if (c == 0) {
      float m2 = red[r][0];
      for (int i = 1; i < 16; ++i) m2 = fmaxf(m2, red[r][i]);
      rowmax[r] = m2;
    }
    __syncthreads();
    float sum = 0.0f;
    for (int i = 0; i < 32; ++i) {
      const int j = c + (i << 4);
      const float e = __expf(scores[r][j] - rowmax[r]);
      scores[r][j] = e;
      sum += e;
    }
    red[r][c] = sum; __syncthreads();
    if (c == 0) {
      float s2 = red[r][0];
      for (int i = 1; i < 16; ++i) s2 += red[r][i];
      rowsum[r] = s2;
    }
    __syncthreads();
    // (1-g)*softmax + g*wpos already row-sums to 1 -> reference renorm is identity.
    const float inv = (1.0f - g) / rowsum[r];
    for (int i = 0; i < 32; ++i) {
      const int j = c + (i << 4);
      pLDS[r][j] = (__bf16)(scores[r][j] * inv + g * wp[j]);
    }
  }
  __syncthreads();

  // ---- Phase 3: out = P @ V, M=16, N=64, K=512, split-K over wave pairs ----
  {
    const int nt = w & 3;
    const int ks = (w >> 2) * 256;
    const __bf16* vcol = vT + ((size_t)bhid * kDH + nt * 16 + (lane & 15)) * kS;
    v8f acc = {};
    for (int k0 = ks; k0 < ks + 256; k0 += 32) {
      v16bf a = load_a_frag(&pLDS[lane & 15][0], k0, lane);
      v16bf bfr = load_b_frag(vcol, k0, lane);
      acc = wmma_bf16(a, bfr, acc);
    }
    if (w >= 4) {
#pragma unroll
      for (int v = 0; v < 8; ++v) partial[nt][v + rofs][lane & 15] = acc[v];
    }
    __syncthreads();
    if (w < 4) {
#pragma unroll
      for (int v = 0; v < 8; ++v) {
        const float val = acc[v] + partial[nt][v + rofs][lane & 15];
        const int s = q0 + v + rofs;
        const int col = h * kDH + nt * 16 + (lane & 15);
        aout[((size_t)b * kS + s) * kDIM + col] = (__bf16)val;
      }
    }
  }
}

// ---------------------------------------------------------------------------
// Output projection: aout[2048,512] @ Wo + bo -> fp32 out.
// ---------------------------------------------------------------------------
__global__ void k_out_gemm(const __bf16* __restrict__ aout, const __bf16* __restrict__ WoT,
                           const float* __restrict__ bo, float* __restrict__ out) {
  const int w = threadIdx.x >> 5, lane = threadIdx.x & 31;
  const int m0 = (blockIdx.y * 8 + w) * 16;
  const int n0 = blockIdx.x * 64;
  const __bf16* arow = aout + (size_t)(m0 + (lane & 15)) * kDIM;

  v8f acc[4] = {};
  for (int k0 = 0; k0 < kDIM; k0 += 32) {
    v16bf a = load_a_frag(arow, k0, lane);
#pragma unroll
    for (int t = 0; t < 4; ++t) {
      v16bf bfr = load_b_frag(WoT + (size_t)(n0 + t * 16 + (lane & 15)) * kDIM, k0, lane);
      acc[t] = wmma_bf16(a, bfr, acc[t]);
    }
  }
  const int rofs = (lane >> 4) << 3;
#pragma unroll
  for (int t = 0; t < 4; ++t)
#pragma unroll
    for (int v = 0; v < 8; ++v) {
      const int m = m0 + v + rofs;
      const int n = n0 + t * 16 + (lane & 15);
      out[(size_t)m * kDIM + n] = acc[t][v] + bo[n];
    }
}

// ---------------------------------------------------------------------------
// Host entry
// ---------------------------------------------------------------------------
extern "C" void kernel_launch(void* const* d_in, const int* in_sizes, int n_in,
                              void* d_out, int out_size, void* d_ws, size_t ws_size,
                              hipStream_t stream) {
  (void)in_sizes; (void)n_in; (void)out_size; (void)ws_size;
  const float* x    = (const float*)d_in[0];
  const float* pos  = (const float*)d_in[1];
  const float* Wq   = (const float*)d_in[2];
  const float* Wk   = (const float*)d_in[3];
  const float* Wv   = (const float*)d_in[4];
  const float* Wo   = (const float*)d_in[5];
  const float* bo   = (const float*)d_in[6];
  const float* Wp1  = (const float*)d_in[7];
  const float* bp1  = (const float*)d_in[8];
  const float* Wp2  = (const float*)d_in[9];
  const float* bp2  = (const float*)d_in[10];
  const float* Wh   = (const float*)d_in[11];
  // d_in[12] = bh : cancels inside the positional softmax (shift invariance)
  const float* gate = (const float*)d_in[13];

  uint8_t* p = (uint8_t*)d_ws;
  auto take = [&](size_t bytes) -> void* {
    void* r = p;
    p += (bytes + 255) & ~(size_t)255;
    return r;
  };
  __bf16* xb   = (__bf16*)take((size_t)kTOK * kDIM * 2);
  __bf16* WqT  = (__bf16*)take((size_t)kDIM * kDIM * 2);
  __bf16* WkT  = (__bf16*)take((size_t)kDIM * kDIM * 2);
  __bf16* WvT  = (__bf16*)take((size_t)kDIM * kDIM * 2);
  __bf16* WoT  = (__bf16*)take((size_t)kDIM * kDIM * 2);
  __bf16* qh   = (__bf16*)take((size_t)kB * kH * kS * kDH * 2);
  __bf16* kh   = (__bf16*)take((size_t)kB * kH * kS * kDH * 2);
  __bf16* vT   = (__bf16*)take((size_t)kB * kH * kDH * kS * 2);
  __bf16* aout = (__bf16*)take((size_t)kTOK * kDIM * 2);
  float*  wpos = (float*)take((size_t)kB * kH * kS * 4);

  k_convert_x<<<(kTOK * kDIM) / 256, 256, 0, stream>>>(x, xb, kTOK * kDIM);
  k_transpose_w<<<dim3((kDIM * kDIM) / 256, 4), 256, 0, stream>>>(
      Wq, Wk, Wv, Wo, WqT, WkT, WvT, WoT);
  k_pos<<<kB * kH, kS, 0, stream>>>(pos, Wp1, bp1, Wp2, bp2, Wh, wpos);
  k_qkv_gemm<<<dim3(8, 16, 3), 256, 0, stream>>>(xb, WqT, WkT, WvT, qh, kh, vT);
  k_attn<<<dim3(kS / 16, kB * kH), 256, 0, stream>>>(qh, kh, vT, wpos, gate, aout);
  k_out_gemm<<<dim3(8, 16), 256, 0, stream>>>(aout, WoT, bo, (float*)d_out);
}
